// PAINN_55370718380133
// MI455X (gfx1250) — compile-verified, matrix operand order checked by reading, not run on hardware
//
#include <hip/hip_runtime.h>

// ---------------------------------------------------------------------------
// PaiNN message-passing network for MI455X (gfx1250, wave32, WMMA bf16).
// ---------------------------------------------------------------------------

typedef __bf16 bf16_t;
typedef __attribute__((ext_vector_type(16))) __bf16 v16bf;
typedef __attribute__((ext_vector_type(8)))  float  v8f;

#define NN_NODES 20000
#define NE_EDGES 320000
#define SDIM     128
#define NGRAPH   32
#define PI_F     3.14159265358979f

// Fast silu: raw v_rcp_f32 instead of the IEEE div_scale/fma correction chain.
__device__ __forceinline__ float silu_f(float x) {
    return x * __builtin_amdgcn_rcpf(1.0f + __expf(-x));
}

__device__ __forceinline__ v8f wmma_bf16(v16bf a, v16bf b, v8f c) {
    return __builtin_amdgcn_wmma_f32_16x16x32_bf16(
        /*neg_a=*/false, a, /*neg_b=*/false, b,
        /*c_mod=*/(short)0, c, /*reuse_a=*/false, /*reuse_b=*/false);
}

// Load a 16x32 bf16 fragment (A-layout; B uses same layout on N-major weight
// copies).  Lane L (<16): row r0+L, K chunks {k0..k0+7, k0+16..k0+23};
// lane L+16: row r0+L, K chunks {k0+8..k0+15, k0+24..k0+31}.
__device__ __forceinline__ v16bf frag_ld(const bf16_t* base, int r0, int k0,
                                         int ldk, int lane) {
    const bf16_t* p = base + (size_t)(r0 + (lane & 15)) * ldk
                           + (k0 + ((lane >> 4) << 3));
    union { uint4 u; bf16_t b[8]; } a0, a1;
    a0.u = *(const uint4*)p;
    a1.u = *(const uint4*)(p + 16);
    v16bf f;
#pragma unroll
    for (int j = 0; j < 8; ++j) { f[j] = a0.b[j]; f[8 + j] = a1.b[j]; }
    return f;
}

// ---------------------------------------------------------------------------
// Prep kernels
// ---------------------------------------------------------------------------
__global__ void k_prep_edges(const float* __restrict__ ev,
                             float* __restrict__ dirs,
                             float* __restrict__ fcm,
                             bf16_t* __restrict__ rbf, int E) {
    int e = blockIdx.x * blockDim.x + threadIdx.x;
    if (e >= E) return;
    float x = ev[e * 3 + 0], y = ev[e * 3 + 1], z = ev[e * 3 + 2];
    float r  = sqrtf(x * x + y * y + z * z);
    float rs = fmaxf(r, 1e-9f);
    float inv = __builtin_amdgcn_rcpf(rs);
    dirs[e * 3 + 0] = x * inv;
    dirs[e * 3 + 1] = y * inv;
    dirs[e * 3 + 2] = z * inv;
    float nbr = (r <= 4.0f) ? 1.0f : 0.0f;
    float fc  = 0.5f * (__cosf(PI_F * r / 2.0f) + 1.0f) * ((r < 2.0f) ? 1.0f : 0.0f);
    fcm[e] = fc * nbr;
    const float w = PI_F / 4.0f;
#pragma unroll
    for (int k = 0; k < 32; ++k) {
        float v = (k < 20) ? (__sinf((float)(k + 1) * w * rs) * inv) : 0.0f;
        rbf[(size_t)e * 32 + k] = (bf16_t)v;
    }
}

// dst[n*ldk+k] = src[k*N+n] (zero-pad k in [K, ldk))
__global__ void k_transpose(const float* __restrict__ src,
                            bf16_t* __restrict__ dst, int K, int N, int ldk) {
    int i = blockIdx.x * blockDim.x + threadIdx.x;
    if (i >= N * ldk) return;
    int n = i / ldk, k = i % ldk;
    dst[(size_t)n * ldk + k] = (bf16_t)((k < K) ? src[(size_t)k * N + n] : 0.0f);
}

__global__ void k_snap_state(const float* __restrict__ s,
                             bf16_t* __restrict__ d, int n) {
    int i = blockIdx.x * blockDim.x + threadIdx.x;
    if (i < n) d[i] = (bf16_t)s[i];
}

__global__ void k_copy_f32(const float* __restrict__ s,
                           float* __restrict__ d, size_t n) {
    size_t i = (size_t)blockIdx.x * blockDim.x + threadIdx.x;
    if (i < n) d[i] = s[i];
}

// ---------------------------------------------------------------------------
// Edge kernel: 128 edges/block, 256 threads (8 waves).
//   h   = silu(state_b[from] @ Wphi1 + b1)           (WMMA, K=128)
//   phi = h @ Wphi2 + b2                             (WMMA, N=384 in 3 chunks)
//   flt = (rbf @ Wfilt + bf) * fcm                   (WMMA, K=32 padded)
//   W = phi*flt ; scatter Ws2 -> state ; dv -> state_vec   (atomics)
// ---------------------------------------------------------------------------
#define E_AS   0
#define E_HS   34816
#define E_WS   69632
#define E_RS   104448
#define E_FS   114688
#define E_SB1  124928
#define E_SB2  125440
#define E_SBF  125952
#define E_FROM 126464
#define E_TO   126976
#define E_FCM  127488
#define E_DIR  128000
#define E_SMEM 129536

__global__ __launch_bounds__(256)
void k_edge(const bf16_t* __restrict__ state_b,
            const float*  __restrict__ vec_old,
            const float*  __restrict__ dirs,
            const float*  __restrict__ fcm,
            const bf16_t* __restrict__ rbf,
            const int*    __restrict__ nfrom,
            const int*    __restrict__ nto,
            const bf16_t* __restrict__ Wphi1T,
            const bf16_t* __restrict__ Wphi2T,
            const bf16_t* __restrict__ WfiltT,
            const float*  __restrict__ b1g,
            const float*  __restrict__ b2g,
            const float*  __restrict__ bfg,
            float* __restrict__ state,
            float* __restrict__ svec) {
    extern __shared__ char smem[];
    bf16_t* As  = (bf16_t*)(smem + E_AS);    // 128 x 136
    bf16_t* Hs  = (bf16_t*)(smem + E_HS);    // 128 x 136
    bf16_t* Ws  = (bf16_t*)(smem + E_WS);    // 128 x 136
    bf16_t* Rs  = (bf16_t*)(smem + E_RS);    // 128 x 40
    bf16_t* Fs  = (bf16_t*)(smem + E_FS);    // 128 x 40
    float*  sb1 = (float*) (smem + E_SB1);
    float*  sb2 = (float*) (smem + E_SB2);
    float*  sbf = (float*) (smem + E_SBF);
    int*    sFr = (int*)   (smem + E_FROM);
    int*    sTo = (int*)   (smem + E_TO);
    float*  sFc = (float*) (smem + E_FCM);
    float*  sDr = (float*) (smem + E_DIR);

    const int tid  = threadIdx.x;
    const int lane = tid & 31;
    const int wv   = tid >> 5;
    const int e0   = blockIdx.x * 128;
    const int m0   = wv * 16;

    // --- stage metadata, Wphi1T, rbf, bias ---
    for (int i = tid; i < 128; i += 256) {
        sFr[i] = nfrom[e0 + i];
        sTo[i] = nto[e0 + i];
        sFc[i] = fcm[e0 + i];
        sDr[i * 3 + 0] = dirs[(size_t)(e0 + i) * 3 + 0];
        sDr[i * 3 + 1] = dirs[(size_t)(e0 + i) * 3 + 1];
        sDr[i * 3 + 2] = dirs[(size_t)(e0 + i) * 3 + 2];
        sb1[i] = b1g[i];
    }
    for (int i = tid; i < 128 * 16; i += 256) {
        int r = i >> 4, c = (i & 15) * 8;
        *(uint4*)(Ws + r * 136 + c) = *(const uint4*)(Wphi1T + (size_t)r * 128 + c);
    }
    for (int i = tid; i < 128 * 4; i += 256) {
        int r = i >> 2, c = (i & 3) * 8;
        *(uint4*)(Rs + r * 40 + c) = *(const uint4*)(rbf + (size_t)(e0 + r) * 32 + c);
    }
    __syncthreads();

    // --- gather s_j (bf16) ---
    for (int i = tid; i < 128 * 16; i += 256) {
        int r = i >> 4, c = (i & 15) * 8;
        *(uint4*)(As + r * 136 + c) =
            *(const uint4*)(state_b + (size_t)sFr[r] * 128 + c);
    }
    __syncthreads();

    // --- GEMM1: h = silu(s_j @ Wphi1 + b1) ---
    v16bf a0 = frag_ld(As, m0, 0,  136, lane);
    v16bf a1 = frag_ld(As, m0, 32, 136, lane);
    v16bf a2 = frag_ld(As, m0, 64, 136, lane);
    v16bf a3 = frag_ld(As, m0, 96, 136, lane);
#pragma unroll
    for (int nt = 0; nt < 8; ++nt) {
        v8f acc = {};
        acc = wmma_bf16(a0, frag_ld(Ws, nt * 16, 0,  136, lane), acc);
        acc = wmma_bf16(a1, frag_ld(Ws, nt * 16, 32, 136, lane), acc);
        acc = wmma_bf16(a2, frag_ld(Ws, nt * 16, 64, 136, lane), acc);
        acc = wmma_bf16(a3, frag_ld(Ws, nt * 16, 96, 136, lane), acc);
        int n  = nt * 16 + (lane & 15);
        int mb = m0 + ((lane >> 4) << 3);
#pragma unroll
        for (int r = 0; r < 8; ++r)
            Hs[(mb + r) * 136 + n] = (bf16_t)silu_f(acc[r] + sb1[n]);
    }
    __syncthreads();

    // --- GEMM2 + filter (3 chunks of N=128) ---
    v16bf h0 = frag_ld(Hs, m0, 0,  136, lane);
    v16bf h1 = frag_ld(Hs, m0, 32, 136, lane);
    v16bf h2 = frag_ld(Hs, m0, 64, 136, lane);
    v16bf h3 = frag_ld(Hs, m0, 96, 136, lane);
    v16bf rf = frag_ld(Rs, m0, 0, 40, lane);
    float w1s[8][8];

#pragma unroll
    for (int c = 0; c < 3; ++c) {
        for (int i = tid; i < 128 * 16; i += 256) {
            int r = i >> 4, cc = (i & 15) * 8;
            *(uint4*)(Ws + r * 136 + cc) =
                *(const uint4*)(Wphi2T + (size_t)(c * 128 + r) * 128 + cc);
        }
        for (int i = tid; i < 128 * 4; i += 256) {
            int r = i >> 2, cc = (i & 3) * 8;
            *(uint4*)(Fs + r * 40 + cc) =
                *(const uint4*)(WfiltT + (size_t)(c * 128 + r) * 32 + cc);
        }
        for (int i = tid; i < 128; i += 256) {
            sb2[i] = b2g[c * 128 + i];
            sbf[i] = bfg[c * 128 + i];
        }
        __syncthreads();

#pragma unroll
        for (int nt = 0; nt < 8; ++nt) {
            v8f acc = {};
            acc = wmma_bf16(h0, frag_ld(Ws, nt * 16, 0,  136, lane), acc);
            acc = wmma_bf16(h1, frag_ld(Ws, nt * 16, 32, 136, lane), acc);
            acc = wmma_bf16(h2, frag_ld(Ws, nt * 16, 64, 136, lane), acc);
            acc = wmma_bf16(h3, frag_ld(Ws, nt * 16, 96, 136, lane), acc);
            v8f fac = {};
            fac = wmma_bf16(rf, frag_ld(Fs, nt * 16, 0, 40, lane), fac);

            int n  = nt * 16 + (lane & 15);
            int mb = m0 + ((lane >> 4) << 3);
#pragma unroll
            for (int r = 0; r < 8; ++r) {
                int m = mb + r;
                float w = (acc[r] + sb2[n]) * ((fac[r] + sbf[n]) * sFc[m]);
                if (c == 0) {
                    w1s[nt][r] = w;                               // Ws1
                } else if (c == 1) {
                    atomicAdd(state + (size_t)sTo[m] * 128 + n, w); // Ws2
                } else {
                    int nf = sFr[m], to = sTo[m];                 // Ws3
#pragma unroll
                    for (int d = 0; d < 3; ++d) {
                        float dv = vec_old[((size_t)nf * 3 + d) * 128 + n] * w1s[nt][r]
                                 + sDr[m * 3 + d] * w;
                        atomicAdd(svec + ((size_t)to * 3 + d) * 128 + n, dv);
                    }
                }
            }
        }
        __syncthreads();
    }
}

// ---------------------------------------------------------------------------
// Node kernel: 64 nodes/block, 128 threads (4 waves).
// ---------------------------------------------------------------------------
#define N_AV   0
#define N_WT   17408
#define N_VS   84992
#define N_CIN  134144
#define N_VN2  167936
#define N_H2   200704
#define N_AVV  218112
#define N_ASV  234496
#define N_SB   267264
#define N_SB2  267776
#define N_SMEM 268288

__global__ __launch_bounds__(128)
void k_node(float* __restrict__ state,
            float* __restrict__ svec,
            const bf16_t* __restrict__ WUT,
            const bf16_t* __restrict__ WVT,
            const bf16_t* __restrict__ Wup1T,
            const bf16_t* __restrict__ Wup2T,
            const float*  __restrict__ bu1,
            const float*  __restrict__ bu2,
            int nNodes) {
    extern __shared__ char smem[];
    bf16_t* Av  = (bf16_t*)(smem + N_AV);    // 64 x 136
    bf16_t* Wt  = (bf16_t*)(smem + N_WT);    // 128 x 264
    bf16_t* Vs  = (bf16_t*)(smem + N_VS);    // 3 x 64 x 128
    bf16_t* Cin = (bf16_t*)(smem + N_CIN);   // 64 x 264
    float*  Vn2 = (float*) (smem + N_VN2);   // 64 x 128
    bf16_t* H2  = (bf16_t*)(smem + N_H2);    // 64 x 136
    bf16_t* Avv = (bf16_t*)(smem + N_AVV);   // 64 x 128
    float*  Asv = (float*) (smem + N_ASV);   // 64 x 128
    float*  sb  = (float*) (smem + N_SB);
    float*  sb2 = (float*) (smem + N_SB2);

    const int tid  = threadIdx.x;
    const int lane = tid & 31;
    const int wv   = tid >> 5;
    const int n0   = blockIdx.x * 64;
    const int m0   = wv * 16;

    for (int i = tid; i < 64 * 128; i += 128) Vn2[i] = 0.0f;
    for (int i = tid; i < 128 * 16; i += 128) {
        int r = i >> 4, c = (i & 15) * 8;
        *(uint4*)(Wt + r * 264 + c) = *(const uint4*)(WVT + (size_t)r * 128 + c);
    }
    __syncthreads();

    // --- Phase 1: V = vec @ W_V ; Vnorm^2 ---
    for (int d = 0; d < 3; ++d) {
        for (int i = tid; i < 64 * 128; i += 128) {
            int r = i >> 7, c = i & 127;
            int nn = n0 + r;
            float v = (nn < nNodes) ? svec[((size_t)nn * 3 + d) * 128 + c] : 0.0f;
            Av[r * 136 + c] = (bf16_t)v;
        }
        __syncthreads();
        v16bf a0 = frag_ld(Av, m0, 0,  136, lane);
        v16bf a1 = frag_ld(Av, m0, 32, 136, lane);
        v16bf a2 = frag_ld(Av, m0, 64, 136, lane);
        v16bf a3 = frag_ld(Av, m0, 96, 136, lane);
#pragma unroll
        for (int nt = 0; nt < 8; ++nt) {
            v8f acc = {};
            acc = wmma_bf16(a0, frag_ld(Wt, nt * 16, 0,  264, lane), acc);
            acc = wmma_bf16(a1, frag_ld(Wt, nt * 16, 32, 264, lane), acc);
            acc = wmma_bf16(a2, frag_ld(Wt, nt * 16, 64, 264, lane), acc);
            acc = wmma_bf16(a3, frag_ld(Wt, nt * 16, 96, 264, lane), acc);
            int n  = nt * 16 + (lane & 15);
            int mb = m0 + ((lane >> 4) << 3);
#pragma unroll
            for (int r = 0; r < 8; ++r) {
                int m = mb + r;
                float v = acc[r];
                Vs[((size_t)d * 64 + m) * 128 + n] = (bf16_t)v;
                Vn2[m * 128 + n] += v * v;
            }
        }
        __syncthreads();
    }

    // --- Phase 2: Cin = [Vnorm, state] ; t = silu(Cin @ Wup1 + b) ; a = t @ Wup2 + b ---
    for (int i = tid; i < 64 * 128; i += 128) {
        int r = i >> 7, c = i & 127;
        Cin[r * 264 + c] = (bf16_t)sqrtf(Vn2[r * 128 + c]);
        int nn = n0 + r;
        float sv = (nn < nNodes) ? state[(size_t)nn * 128 + c] : 0.0f;
        Cin[r * 264 + 128 + c] = (bf16_t)sv;
    }
    for (int i = tid; i < 128 * 32; i += 128) {
        int r = i >> 5, c = (i & 31) * 8;
        *(uint4*)(Wt + r * 264 + c) = *(const uint4*)(Wup1T + (size_t)r * 256 + c);
    }
    for (int i = tid; i < 128; i += 128) sb[i] = bu1[i];
    __syncthreads();

    {
        v16bf cf[8];
#pragma unroll
        for (int kt = 0; kt < 8; ++kt) cf[kt] = frag_ld(Cin, m0, kt * 32, 264, lane);
#pragma unroll
        for (int nt = 0; nt < 8; ++nt) {
            v8f acc = {};
#pragma unroll
            for (int kt = 0; kt < 8; ++kt)
                acc = wmma_bf16(cf[kt], frag_ld(Wt, nt * 16, kt * 32, 264, lane), acc);
            int n  = nt * 16 + (lane & 15);
            int mb = m0 + ((lane >> 4) << 3);
#pragma unroll
            for (int r = 0; r < 8; ++r)
                H2[(mb + r) * 136 + n] = (bf16_t)silu_f(acc[r] + sb[n]);
        }
    }
    __syncthreads();

    v16bf hf0 = frag_ld(H2, m0, 0,  136, lane);
    v16bf hf1 = frag_ld(H2, m0, 32, 136, lane);
    v16bf hf2 = frag_ld(H2, m0, 64, 136, lane);
    v16bf hf3 = frag_ld(H2, m0, 96, 136, lane);

#pragma unroll
    for (int c = 0; c < 3; ++c) {
        for (int i = tid; i < 128 * 16; i += 128) {
            int r = i >> 4, cc = (i & 15) * 8;
            *(uint4*)(Wt + r * 264 + cc) =
                *(const uint4*)(Wup2T + (size_t)(c * 128 + r) * 128 + cc);
        }
        for (int i = tid; i < 128; i += 128) sb2[i] = bu2[c * 128 + i];
        __syncthreads();
#pragma unroll
        for (int nt = 0; nt < 8; ++nt) {
            v8f acc = {};
            acc = wmma_bf16(hf0, frag_ld(Wt, nt * 16, 0,  264, lane), acc);
            acc = wmma_bf16(hf1, frag_ld(Wt, nt * 16, 32, 264, lane), acc);
            acc = wmma_bf16(hf2, frag_ld(Wt, nt * 16, 64, 264, lane), acc);
            acc = wmma_bf16(hf3, frag_ld(Wt, nt * 16, 96, 264, lane), acc);
            int n  = nt * 16 + (lane & 15);
            int mb = m0 + ((lane >> 4) << 3);
#pragma unroll
            for (int r = 0; r < 8; ++r) {
                int m = mb + r, nn = n0 + m;
                float val = acc[r] + sb2[n];
                if (c == 0) Avv[m * 128 + n] = (bf16_t)val;     // a_vv
                else if (c == 1) Asv[m * 128 + n] = val;        // a_sv
                else if (nn < nNodes)                           // a_ss
                    state[(size_t)nn * 128 + n] += val;
            }
        }
        __syncthreads();
    }

    // --- Phase 3: U = vec @ W_U ; uv_dot ; state_vec += U * a_vv ---
    for (int i = tid; i < 128 * 16; i += 128) {
        int r = i >> 4, c = (i & 15) * 8;
        *(uint4*)(Wt + r * 264 + c) = *(const uint4*)(WUT + (size_t)r * 128 + c);
    }
    __syncthreads();

    float uv[8][8];
#pragma unroll
    for (int nt = 0; nt < 8; ++nt)
#pragma unroll
        for (int r = 0; r < 8; ++r) uv[nt][r] = 0.0f;

    for (int d = 0; d < 3; ++d) {
        for (int i = tid; i < 64 * 128; i += 128) {
            int r = i >> 7, c = i & 127;
            int nn = n0 + r;
            float v = (nn < nNodes) ? svec[((size_t)nn * 3 + d) * 128 + c] : 0.0f;
            Av[r * 136 + c] = (bf16_t)v;
        }
        __syncthreads();
        v16bf a0 = frag_ld(Av, m0, 0,  136, lane);
        v16bf a1 = frag_ld(Av, m0, 32, 136, lane);
        v16bf a2 = frag_ld(Av, m0, 64, 136, lane);
        v16bf a3 = frag_ld(Av, m0, 96, 136, lane);
#pragma unroll
        for (int nt = 0; nt < 8; ++nt) {
            v8f acc = {};
            acc = wmma_bf16(a0, frag_ld(Wt, nt * 16, 0,  264, lane), acc);
            acc = wmma_bf16(a1, frag_ld(Wt, nt * 16, 32, 264, lane), acc);
            acc = wmma_bf16(a2, frag_ld(Wt, nt * 16, 64, 264, lane), acc);
            acc = wmma_bf16(a3, frag_ld(Wt, nt * 16, 96, 264, lane), acc);
            int n  = nt * 16 + (lane & 15);
            int mb = m0 + ((lane >> 4) << 3);
#pragma unroll
            for (int r = 0; r < 8; ++r) {
                int m = mb + r, nn = n0 + m;
                float u = acc[r];
                uv[nt][r] += u * (float)Vs[((size_t)d * 64 + m) * 128 + n];
                if (nn < nNodes)
                    svec[((size_t)nn * 3 + d) * 128 + n] += u * (float)Avv[m * 128 + n];
            }
        }
        __syncthreads();
    }

    // --- Phase 4: state += a_sv * uv_dot ---
#pragma unroll
    for (int nt = 0; nt < 8; ++nt) {
        int n  = nt * 16 + (lane & 15);
        int mb = m0 + ((lane >> 4) << 3);
#pragma unroll
        for (int r = 0; r < 8; ++r) {
            int m = mb + r, nn = n0 + m;
            if (nn < nNodes)
                state[(size_t)nn * 128 + n] += Asv[m * 128 + n] * uv[nt][r];
        }
    }
}

// ---------------------------------------------------------------------------
// Readout
// ---------------------------------------------------------------------------
__global__ void k_graph_sum(const float* __restrict__ state,
                            const int* __restrict__ gidx,
                            float* __restrict__ gs, int nNodes) {
    int i = blockIdx.x * blockDim.x + threadIdx.x;
    if (i >= nNodes * 128) return;
    int node = i >> 7, s = i & 127;
    atomicAdd(&gs[gidx[node] * 128 + s], state[i]);
}

__global__ void k_out(const float* __restrict__ gs,
                      const float* __restrict__ W1,
                      const float* __restrict__ b1,
                      const float* __restrict__ W2,
                      const float* __restrict__ b2,
                      float* __restrict__ out) {
    int g = blockIdx.x, t = threadIdx.x;
    float acc = b1[t];
    for (int s = 0; s < 128; ++s) acc += gs[g * 128 + s] * W1[s * 128 + t];
    float h = silu_f(acc) * W2[t];
    __shared__ float red[128];
    red[t] = h;
    __syncthreads();
    for (int off = 64; off > 0; off >>= 1) {
        if (t < off) red[t] += red[t + off];
        __syncthreads();
    }
    if (t == 0) out[g] = red[0] + b2[0];
}

// ---------------------------------------------------------------------------
// Launch
// ---------------------------------------------------------------------------
extern "C" void kernel_launch(void* const* d_in, const int* in_sizes, int n_in,
                              void* d_out, int out_size, void* d_ws, size_t ws_size,
                              hipStream_t stream) {
    (void)in_sizes; (void)n_in; (void)out_size; (void)ws_size;
    const float* ev     = (const float*)d_in[0];
    const int*   nfrom  = (const int*)  d_in[1];
    const int*   nto    = (const int*)  d_in[2];
    const int*   gidx   = (const int*)  d_in[3];
    const float* W_phi1 = (const float*)d_in[4];
    const float* b_phi1 = (const float*)d_in[5];
    const float* W_phi2 = (const float*)d_in[6];
    const float* b_phi2 = (const float*)d_in[7];
    const float* W_filt = (const float*)d_in[8];
    const float* b_filt = (const float*)d_in[9];
    const float* W_U    = (const float*)d_in[10];
    const float* W_V    = (const float*)d_in[11];
    const float* W_up1  = (const float*)d_in[12];
    const float* b_up1  = (const float*)d_in[13];
    const float* W_up2  = (const float*)d_in[14];
    const float* b_up2  = (const float*)d_in[15];
    const float* W_out1 = (const float*)d_in[16];
    const float* b_out1 = (const float*)d_in[17];
    const float* W_out2 = (const float*)d_in[18];
    const float* b_out2 = (const float*)d_in[19];
    float* out = (float*)d_out;

    const int N = NN_NODES, E = NE_EDGES, S = SDIM, G = NGRAPH;

    char* p = (char*)d_ws;
    auto take = [&](size_t bytes) -> char* {
        char* q = p;
        p += (bytes + 255) & ~(size_t)255;
        return q;
    };
    float*  state   = (float*) take((size_t)N * S * 4);
    float*  svec    = (float*) take((size_t)N * 3 * S * 4);
    bf16_t* state_b = (bf16_t*)take((size_t)N * S * 2);
    float*  vec_old = (float*) take((size_t)N * 3 * S * 4);
    float*  dirs    = (float*) take((size_t)E * 3 * 4);
    float*  fcm     = (float*) take((size_t)E * 4);
    bf16_t* rbf     = (bf16_t*)take((size_t)E * 32 * 2);
    bf16_t* Wphi1T  = (bf16_t*)take((size_t)S * S * 2);
    bf16_t* Wphi2T  = (bf16_t*)take((size_t)3 * S * S * 2);
    bf16_t* WfiltT  = (bf16_t*)take((size_t)384 * 32 * 2);
    bf16_t* WUT     = (bf16_t*)take((size_t)S * S * 2);
    bf16_t* WVT     = (bf16_t*)take((size_t)S * S * 2);
    bf16_t* Wup1T   = (bf16_t*)take((size_t)S * 2 * S * 2);
    bf16_t* Wup2T   = (bf16_t*)take((size_t)3 * S * S * 2);
    float*  gstate  = (float*) take((size_t)G * S * 4);

    hipMemsetAsync(state, 0, (size_t)N * S * 4, stream);
    hipMemsetAsync(svec, 0, (size_t)N * 3 * S * 4, stream);
    hipMemsetAsync(gstate, 0, (size_t)G * S * 4, stream);

    k_prep_edges<<<(E + 255) / 256, 256, 0, stream>>>(ev, dirs, fcm, rbf, E);
    k_transpose<<<(S * S + 255) / 256, 256, 0, stream>>>(W_phi1, Wphi1T, S, S, S);
    k_transpose<<<(3 * S * S + 255) / 256, 256, 0, stream>>>(W_phi2, Wphi2T, S, 3 * S, S);
    k_transpose<<<(384 * 32 + 255) / 256, 256, 0, stream>>>(W_filt, WfiltT, 20, 384, 32);
    k_transpose<<<(S * S + 255) / 256, 256, 0, stream>>>(W_U, WUT, S, S, S);
    k_transpose<<<(S * S + 255) / 256, 256, 0, stream>>>(W_V, WVT, S, S, S);
    k_transpose<<<(S * 2 * S + 255) / 256, 256, 0, stream>>>(W_up1, Wup1T, 2 * S, S, 2 * S);
    k_transpose<<<(3 * S * S + 255) / 256, 256, 0, stream>>>(W_up2, Wup2T, S, 3 * S, S);

    for (int round = 0; round < 3; ++round) {
        k_snap_state<<<(N * S + 255) / 256, 256, 0, stream>>>(state, state_b, N * S);
        k_copy_f32<<<(int)(((size_t)N * 3 * S + 255) / 256), 256, 0, stream>>>(
            svec, vec_old, (size_t)N * 3 * S);
        k_edge<<<E / 128, 256, E_SMEM, stream>>>(
            state_b, vec_old, dirs, fcm, rbf, nfrom, nto,
            Wphi1T, Wphi2T, WfiltT, b_phi1, b_phi2, b_filt, state, svec);
        k_node<<<(N + 63) / 64, 128, N_SMEM, stream>>>(
            state, svec, WUT, WVT, Wup1T, Wup2T, b_up1, b_up2, N);
    }

    k_graph_sum<<<(N * S + 255) / 256, 256, 0, stream>>>(state, gidx, gstate, N);
    k_out<<<G, 128, 0, stream>>>(gstate, W_out1, b_out1, W_out2, b_out2, out);
}
